// GATv2ResidualNet_12549894439332
// MI455X (gfx1250) — matrix operand order repaired and checked by reading, not compiled, not run on hardware
//
#include <hip/hip_runtime.h>
#include <hip/hip_bf16.h>

// ---------------------------------------------------------------------------
// GATv2 residual net for MI455X (gfx1250, wave32, WMMA).
// - GEMMs: v_wmma_f32_16x16x32_bf16, A-strip staged in LDS via TDM
//   (tensor_load_to_lds + s_wait_tensorcnt), B read as b128 from global.
// - Edge softmax/aggregation: f32 atomics (node tables fit in 192MB L2).
// ---------------------------------------------------------------------------

#define N_NODES 100000
#define N_EDGES 1600000
#define NHEADS  8
#define KP_MAX  224

typedef __attribute__((ext_vector_type(16))) __bf16 bf16x16;
typedef __attribute__((ext_vector_type(8)))  float  f32x8;
typedef unsigned int u32x4 __attribute__((ext_vector_type(4)));
typedef int          i32x8 __attribute__((ext_vector_type(8)));
typedef int          i32x4 __attribute__((ext_vector_type(4)));

#if defined(__gfx1250__) && __has_builtin(__builtin_amdgcn_tensor_load_to_lds)
#define USE_TDM 1
#else
#define USE_TDM 0
#endif

// ---------------- helpers ----------------

__device__ __forceinline__ unsigned short f32_to_bf16_bits(float f) {
    unsigned u = __float_as_uint(f);
    u += 0x7FFFu + ((u >> 16) & 1u);   // round-to-nearest-even
    return (unsigned short)(u >> 16);
}

__device__ __forceinline__ unsigned f2key(float f) {
    unsigned u = __float_as_uint(f);
    return (u & 0x80000000u) ? ~u : (u | 0x80000000u);   // order-preserving
}
__device__ __forceinline__ float key2f(unsigned k) {
    unsigned u = (k & 0x80000000u) ? (k ^ 0x80000000u) : ~k;
    return __uint_as_float(u);
}

// ---------------- conversion kernels ----------------

// X[M x K] f32 row-major -> Xb[M x Kp] bf16 row-major, zero-padded K..Kp
__global__ void cvt_feat(const float* __restrict__ X, unsigned short* __restrict__ Xb,
                         int M, int K, int Kp) {
    long long idx = (long long)blockIdx.x * blockDim.x + threadIdx.x;
    if (idx >= (long long)M * Kp) return;
    int r = (int)(idx / Kp);
    int c = (int)(idx % Kp);
    Xb[idx] = (c < K) ? f32_to_bf16_bits(X[(long long)r * K + c]) : (unsigned short)0;
}

// W[fin x N] f32 row-major -> Wt[N x Kp] bf16 row-major (transposed, padded)
__global__ void cvt_weight_t(const float* __restrict__ W, unsigned short* __restrict__ Wt,
                             int fin, int Kp, int Ncols) {
    long long idx = (long long)blockIdx.x * blockDim.x + threadIdx.x;
    if (idx >= (long long)Ncols * Kp) return;
    int col = (int)(idx / Kp);
    int k   = (int)(idx % Kp);
    Wt[idx] = (k < fin) ? f32_to_bf16_bits(W[(long long)k * Ncols + col]) : (unsigned short)0;
}

// ---------------- WMMA GEMM ----------------
// C[M x N] = A[M x Kp]_bf16 @ W[Kp x N]_bf16 (W transposed: Wt[N x Kp]) + bias.
// block = 32 * (N/16) threads; wave w computes column tile w; grid.x = M/16.
// The 16-row A strip (16 x Kp bf16, <= 7KB) is staged in LDS once per block:
// TDM DMA (tensor_load_to_lds) when available, cooperative copy otherwise.

union AFrag { bf16x16 v; uint4 q[2]; };

__global__ void gemm_bf16_wmma(const unsigned short* __restrict__ A,
                               const unsigned short* __restrict__ Wt,
                               const float* __restrict__ bias,
                               float* __restrict__ C,
                               int Kp, int Ncols) {
    __shared__ __align__(16) unsigned short lds_a[16 * KP_MAX];

    const int r0 = blockIdx.x * 16;
    const unsigned short* Astrip = A + (size_t)r0 * Kp;

#if USE_TDM
    if (threadIdx.x < 32) {              // uniform per-wave branch: only wave 0 issues DMA
        // LDS flat address: low 32 bits are the LDS byte offset (aperture rule)
        unsigned lds_off = (unsigned)(unsigned long long)(uintptr_t)&lds_a[0];
        unsigned long long ga = (unsigned long long)(uintptr_t)Astrip;
        // D# group 0: count=1 | lds_addr | global_addr[56:0] | type=2
        u32x4 g0 = { 1u,
                     lds_off,
                     (unsigned)ga,
                     ((unsigned)((ga >> 32) & 0x01FFFFFFull)) | 0x80000000u };
        // D# group 1: data_size=1(2B); tensor_dim0=Kp; tensor_dim1=16;
        //             tile_dim0=Kp; tile_dim1=16; tensor_dim0_stride=Kp
        i32x8 g1 = { (int)(1u << 16),                       // data_size @ [17:16]
                     (int)(((unsigned)Kp & 0xFFFFu) << 16), // tensor_dim0[15:0] @ [63:48]
                     (int)(16u << 16),                      // tensor_dim1[15:0] @ [95:80]->word2[31:16]
                     (int)((unsigned)Kp << 16),             // tile_dim0 @ [127:112]
                     16,                                    // tile_dim1 @ [143:128]
                     (int)Kp,                               // tensor_dim0_stride[31:0] @ [191:160]
                     0, 0 };
        i32x4 z4 = { 0, 0, 0, 0 };
#if __clang_major__ >= 23
        i32x8 z8 = { 0, 0, 0, 0, 0, 0, 0, 0 };
        __builtin_amdgcn_tensor_load_to_lds(g0, g1, z4, z4, z8, 0);
#else
        __builtin_amdgcn_tensor_load_to_lds(g0, g1, z4, z4, 0);
#endif
#if __has_builtin(__builtin_amdgcn_s_wait_tensorcnt)
        __builtin_amdgcn_s_wait_tensorcnt(0);
#endif
    }
    __syncthreads();
#else
    // fallback: cooperative global->LDS copy (b128 both sides)
    {
        const int n16 = (16 * Kp) >> 3;              // uint4 elements in strip
        for (int i = threadIdx.x; i < n16; i += blockDim.x)
            ((uint4*)lds_a)[i] = ((const uint4*)Astrip)[i];
        __syncthreads();
    }
#endif

    const int wave = threadIdx.x >> 5;
    const int lane = threadIdx.x & 31;
    const int c0   = wave * 16;
    const int m    = lane & 15;       // row (A) / col (B,C)
    const int half = lane >> 4;

    const unsigned short* Arow = lds_a + m * Kp;                 // LDS (ds_load_b128)
    const unsigned short* Brow = Wt + (size_t)(c0 + m) * Kp;     // global

    f32x8 acc = {};
    for (int k0 = 0; k0 < Kp; k0 += 32) {
        AFrag fa, fb;
        // A: 16 contiguous K values per lane: K = 16*half + 0..15
        fa.q[0] = *(const uint4*)(Arow + k0 + 16 * half);
        fa.q[1] = *(const uint4*)(Arow + k0 + 16 * half + 8);
        // B: 16 contiguous K values per lane: K = 16*half + 0..15
        fb.q[0] = *(const uint4*)(Brow + k0 + 16 * half);
        fb.q[1] = *(const uint4*)(Brow + k0 + 16 * half + 8);
        acc = __builtin_amdgcn_wmma_f32_16x16x32_bf16(
                  false, fa.v, false, fb.v, (short)0, acc, false, false);
    }

    const int col = c0 + m;
    const float bv = bias ? bias[col] : 0.0f;
#pragma unroll
    for (int r = 0; r < 8; ++r) {
        int orow = r0 + r + 8 * half;
        C[(size_t)orow * Ncols + col] = acc[r] + bv;
    }
}

// ---------------- edge kernels ----------------

// pass 1: e[edge,h] = sum_c leakyrelu(xl[src]+xr[dst]) * att ; segment-max into emax
__global__ void edge_scores(const int* __restrict__ src, const int* __restrict__ dst,
                            const float* __restrict__ xl, const float* __restrict__ xr,
                            const float* __restrict__ att,
                            float* __restrict__ escore, unsigned* __restrict__ emax,
                            int E, int C) {
    int t = blockIdx.x * blockDim.x + threadIdx.x;
    if (t >= E * NHEADS) return;
    int e = t / NHEADS, h = t % NHEADS;
    int s = src[e], d = dst[e];
    int HC = NHEADS * C;
    const float* pl = xl + (size_t)s * HC + h * C;
    const float* pr = xr + (size_t)d * HC + h * C;
    const float* pa = att + h * C;
    float sc = 0.0f;
    for (int c = 0; c < C; ++c) {
        float mv = pl[c] + pr[c];
        mv = (mv > 0.0f) ? mv : 0.2f * mv;        // leaky relu, NEG_SLOPE=0.2
        sc += mv * pa[c];
    }
    escore[t] = sc;
    atomicMax(&emax[(size_t)d * NHEADS + h], f2key(sc));
}

// pass 2: p = exp(e - emax[dst]); denom[dst] += p; acc[dst] += p * xl[src]
__global__ void edge_accum(const int* __restrict__ src, const int* __restrict__ dst,
                           const float* __restrict__ xl,
                           const float* __restrict__ escore,
                           const unsigned* __restrict__ emax,
                           float* __restrict__ denom, float* __restrict__ acc,
                           int E, int C) {
    int t = blockIdx.x * blockDim.x + threadIdx.x;
    if (t >= E * NHEADS) return;
    int e = t / NHEADS, h = t % NHEADS;
    int s = src[e], d = dst[e];
    int HC = NHEADS * C;
    float mx = key2f(emax[(size_t)d * NHEADS + h]);
    float p  = expf(escore[t] - mx);
    atomicAdd(&denom[(size_t)d * NHEADS + h], p);
    const float* pl = xl + (size_t)s * HC + h * C;
    float*       pd = acc + (size_t)d * HC + h * C;
    for (int c = 0; c < C; ++c) atomicAdd(&pd[c], p * pl[c]);
}

// node-level: g = acc/denom + bias; h = gelu_exact(g + res)
__global__ void node_finalize(const float* __restrict__ acc, const float* __restrict__ denom,
                              const float* __restrict__ bias, const float* __restrict__ res,
                              float* __restrict__ out, int C) {
    int HC = NHEADS * C;
    int idx = blockIdx.x * blockDim.x + threadIdx.x;
    if (idx >= N_NODES * HC) return;
    int n = idx / HC, ch = idx % HC, h = ch / C;
    float dn = denom[(size_t)n * NHEADS + h];
    float g  = acc[idx] / (dn + 1e-16f) + bias[ch];
    float xv = g + res[idx];
    out[idx] = 0.5f * xv * (1.0f + erff(xv * 0.70710678118654752f));
}

// ---------------- host orchestration ----------------

static inline int ceil_div(long long a, int b) { return (int)((a + b - 1) / b); }

extern "C" void kernel_launch(void* const* d_in, const int* in_sizes, int n_in,
                              void* d_out, int out_size, void* d_ws, size_t ws_size,
                              hipStream_t stream) {
    (void)n_in; (void)out_size; (void)ws_size;
    const float* x  = (const float*)d_in[0];
    const int*   ei = (const int*)d_in[1];
    const int E = in_sizes[1] / 2;
    const int* src = ei;
    const int* dst = ei + E;

    const float* Wl[4]; const float* bl[4]; const float* Wr[4]; const float* br[4];
    const float* att[4]; const float* bias[4];
    for (int li = 0; li < 4; ++li) {
        Wl[li]   = (const float*)d_in[2 + 6 * li + 0];
        bl[li]   = (const float*)d_in[2 + 6 * li + 1];
        Wr[li]   = (const float*)d_in[2 + 6 * li + 2];
        br[li]   = (const float*)d_in[2 + 6 * li + 3];
        att[li]  = (const float*)d_in[2 + 6 * li + 4];
        bias[li] = (const float*)d_in[2 + 6 * li + 5];
    }
    const float* proj1_W = (const float*)d_in[26];
    const float* proj1_b = (const float*)d_in[27];
    const float* proj4_W = (const float*)d_in[28];
    const float* proj4_b = (const float*)d_in[29];
    const float* lin1_W  = (const float*)d_in[30];
    const float* lin1_b  = (const float*)d_in[31];

    static const int FIN[4] = {193, 128, 128, 128};
    static const int KP [4] = {224, 128, 128, 128};
    static const int HC [4] = {128, 128, 128, 64};
    static const int CC [4] = {16, 16, 16, 8};

    // ---- bump allocator over workspace ----
    char* p = (char*)d_ws;
    auto alloc = [&](size_t bytes) -> void* {
        void* r = (void*)p;
        p += (bytes + 255) & ~(size_t)255;
        return r;
    };
    unsigned short* wtL[4]; unsigned short* wtR[4];
    for (int li = 0; li < 4; ++li) {
        wtL[li] = (unsigned short*)alloc((size_t)HC[li] * KP[li] * 2);
        wtR[li] = (unsigned short*)alloc((size_t)HC[li] * KP[li] * 2);
    }
    unsigned short* wtP1  = (unsigned short*)alloc((size_t)128 * 224 * 2);
    unsigned short* wtP4  = (unsigned short*)alloc((size_t)64 * 128 * 2);
    unsigned short* wtFin = (unsigned short*)alloc((size_t)32 * 64 * 2);
    unsigned short* hbf   = (unsigned short*)alloc((size_t)N_NODES * KP_MAX * 2);
    float* hcur  = (float*)alloc((size_t)N_NODES * 128 * 4);
    float* xl    = (float*)alloc((size_t)N_NODES * 128 * 4);
    float* xr    = (float*)alloc((size_t)N_NODES * 128 * 4);
    float* res   = (float*)alloc((size_t)N_NODES * 128 * 4);
    float* accB  = (float*)alloc((size_t)N_NODES * 128 * 4);
    float* esc   = (float*)alloc((size_t)E * NHEADS * 4);
    unsigned* emax = (unsigned*)alloc((size_t)N_NODES * NHEADS * 4);
    float* denom = (float*)alloc((size_t)N_NODES * NHEADS * 4);

    const int TPB = 256;

    // ---- convert all weights once (transposed + K-padded bf16) ----
    for (int li = 0; li < 4; ++li) {
        long long n = (long long)HC[li] * KP[li];
        cvt_weight_t<<<ceil_div(n, TPB), TPB, 0, stream>>>(Wl[li], wtL[li], FIN[li], KP[li], HC[li]);
        cvt_weight_t<<<ceil_div(n, TPB), TPB, 0, stream>>>(Wr[li], wtR[li], FIN[li], KP[li], HC[li]);
    }
    cvt_weight_t<<<ceil_div((long long)128 * 224, TPB), TPB, 0, stream>>>(proj1_W, wtP1, 193, 224, 128);
    cvt_weight_t<<<ceil_div((long long)64 * 128, TPB), TPB, 0, stream>>>(proj4_W, wtP4, 128, 128, 64);
    cvt_weight_t<<<ceil_div((long long)32 * 64, TPB), TPB, 0, stream>>>(lin1_W, wtFin, 64, 64, 32);

    const int gemm_grid = N_NODES / 16;   // 6250, M divisible by 16

    for (int li = 0; li < 4; ++li) {
        const int fin = FIN[li], kp = KP[li], hc = HC[li], cc = CC[li];
        const float* hin = (li == 0) ? x : hcur;

        // features -> padded bf16
        cvt_feat<<<ceil_div((long long)N_NODES * kp, TPB), TPB, 0, stream>>>(hin, hbf, N_NODES, fin, kp);

        // xl = h@Wl+bl ; xr = h@Wr+br  (WMMA)
        gemm_bf16_wmma<<<gemm_grid, 32 * (hc / 16), 0, stream>>>(hbf, wtL[li], bl[li], xl, kp, hc);
        gemm_bf16_wmma<<<gemm_grid, 32 * (hc / 16), 0, stream>>>(hbf, wtR[li], br[li], xr, kp, hc);

        // residual path
        const float* resptr;
        if (li == 0) {
            gemm_bf16_wmma<<<gemm_grid, 32 * (128 / 16), 0, stream>>>(hbf, wtP1, proj1_b, res, kp, 128);
            resptr = res;
        } else if (li == 3) {
            gemm_bf16_wmma<<<gemm_grid, 32 * (64 / 16), 0, stream>>>(hbf, wtP4, proj4_b, res, kp, 64);
            resptr = res;
        } else {
            resptr = hcur;   // identity residual
        }

        // zero accumulators
        hipMemsetAsync(emax,  0, (size_t)N_NODES * NHEADS * 4, stream);
        hipMemsetAsync(denom, 0, (size_t)N_NODES * NHEADS * 4, stream);
        hipMemsetAsync(accB,  0, (size_t)N_NODES * hc * 4, stream);

        // edge softmax-aggregate (two passes; normalization folded into finalize)
        int eh_blocks = ceil_div((long long)E * NHEADS, TPB);
        edge_scores<<<eh_blocks, TPB, 0, stream>>>(src, dst, xl, xr, att[li], esc, emax, E, cc);
        edge_accum <<<eh_blocks, TPB, 0, stream>>>(src, dst, xl, esc, emax, denom, accB, E, cc);

        // h = gelu(acc/denom + bias + residual)   (writes hcur, stride hc)
        node_finalize<<<ceil_div((long long)N_NODES * hc, TPB), TPB, 0, stream>>>(
            accB, denom, bias[li], resptr, hcur, cc);
    }

    // final linear: [N,64] @ [64,32] + b  -> d_out (f32)
    cvt_feat<<<ceil_div((long long)N_NODES * 64, TPB), TPB, 0, stream>>>(hcur, hbf, N_NODES, 64, 64);
    gemm_bf16_wmma<<<gemm_grid, 32 * (32 / 16), 0, stream>>>(hbf, wtFin, lin1_b, (float*)d_out, 64, 32);
}
